// CRNNTransition_24318104830132
// MI455X (gfx1250) — matrix operand order repaired and checked by reading, not compiled, not run on hardware
//
#include <hip/hip_runtime.h>
#include <cstdint>
#include <cstddef>

// ---------------------------------------------------------------------------
// ConvLSTM (CRNN transition) for MI455X / gfx1250.
// Conv 3x3 -> im2col GEMM: M=512, N=1568, K=2304 via v_wmma_f32_16x16x32_f16.
// Linear head: M=512, N=32, K=6272 with fused bias+ReLU, transposed store.
// GEMM uses register-staged double-buffered LDS: 1 syncthreads per k-tile,
// b128 global loads, b128 LDS fragment loads.
// ---------------------------------------------------------------------------

typedef __attribute__((ext_vector_type(16))) _Float16 v16h;
typedef __attribute__((ext_vector_type(8)))  float    v8f;

#define T_STEPS 32
#define BATCH   32
#define CCH     128          // C
#define HSZ     512          // HS
#define RDI     3            // RD
#define HWD     7
#define SP      49           // 7*7
#define M4C     512          // 4*C  (conv output channels)
#define KCONV   2304         // 2*C*9
#define NCONV   1568         // B*49
#define KLIN    6272         // C*49

#define BM 128
#define BN 64
#define BK 32
#define LDK 40               // padded LDS K-stride (f16 units), keeps 16B align

union V16 { v16h v; uint4 q[2]; };
union U4F16 { uint4 q; _Float16 f[8]; };

// ---------------------------------------------------------------------------
// Generic f16 WMMA GEMM: D[M,N] = A[M,K] * B[K,N] + bias[M]
// flags bit0 = ReLU, bit1 = transposed store D[n*ldd + m] (else D[m*ldd + n]).
// Requirements: M % 128 == 0, K % 32 == 0, N % 8 == 0 (guarded), blockN % 64.
// ---------------------------------------------------------------------------
__global__ __launch_bounds__(256) void gemm_f16_wmma(
    const _Float16* __restrict__ A, const _Float16* __restrict__ Bm,
    float* __restrict__ D, const float* __restrict__ bias,
    int M, int N, int K, int ldd, int flags)
{
    __shared__ __align__(16) _Float16 As[2][BM * LDK];
    __shared__ __align__(16) _Float16 Bs[2][BN * LDK];

    const int tid  = threadIdx.x;
    const int lane = tid & 31;
    const int wave = tid >> 5;
    const int wm   = wave >> 1;      // 0..3 : wave row  (32 rows of M each)
    const int wn   = wave & 1;       // 0..1 : wave col  (32 cols of N each)
    const int lr   = lane & 15;
    const int lh   = lane >> 4;
    const int blockM = blockIdx.y * BM;
    const int blockN = blockIdx.x * BN;

    // --- static staging coordinates (256 threads cover tiles exactly) ---
    const int arow = tid >> 2;            // 0..63   (A rows, +64 for 2nd load)
    const int aseg = (tid & 3) << 3;      // 0,8,16,24 (f16 k-offset)
    const int brow = tid >> 3;            // 0..31   (B k-row)
    const int bseg = (tid & 7) << 3;      // 0..56   (f16 n-offset)
    const int bgn  = blockN + bseg;
    const bool bok = (bgn < N);           // N % 8 == 0 -> whole b128 in/out

    uint4 ra0, ra1, rb;

    auto stage_load = [&](int k0) {
        const _Float16* pa = A + (size_t)(blockM + arow) * K + (k0 + aseg);
        ra0 = *(const uint4*)pa;
        ra1 = *(const uint4*)(pa + (size_t)64 * K);
        uint4 z = {0u, 0u, 0u, 0u};
        rb = z;
        if (bok) rb = *(const uint4*)(Bm + (size_t)(k0 + brow) * N + bgn);
    };
    auto stage_store = [&](int buf) {
        *(uint4*)(&As[buf][arow * LDK + aseg])        = ra0;
        *(uint4*)(&As[buf][(arow + 64) * LDK + aseg]) = ra1;
        U4F16 cv; cv.q = rb;
#pragma unroll
        for (int j = 0; j < 8; ++j)           // transpose: Bs[n][k]
            Bs[buf][(bseg + j) * LDK + brow] = cv.f[j];
    };

    v8f acc[2][2];
    v8f zero = {};
    acc[0][0] = zero; acc[0][1] = zero; acc[1][0] = zero; acc[1][1] = zero;

    const int KT = K / BK;
    stage_load(0);
    stage_store(0);
    __syncthreads();

    for (int kt = 0; kt < KT; ++kt) {
        const int buf = kt & 1;
        if (kt + 1 < KT) stage_load((kt + 1) * BK);   // overlap with compute

        // ---- fragments from LDS (ds_load_b128 pairs) ----
        // A (16x32 f16): lane=(m=lr, half=lh); V0..3 K=2v,2v+1 (+8*half),
        // V4..7 K=16+2(v-4) (+8*half): two contiguous runs of 8 f16.
        V16 af[2], bf[2];
#pragma unroll
        for (int i = 0; i < 2; ++i) {
            const _Float16* p = &As[buf][(wm * 32 + i * 16 + lr) * LDK + lh * 8];
            af[i].q[0] = *(const uint4*)p;
            af[i].q[1] = *(const uint4*)(p + 16);
        }
        // B (32x16 f16): lane=(n=lr, half=lh); same K order as A (two runs of 8).
#pragma unroll
        for (int j = 0; j < 2; ++j) {
            const _Float16* p = &Bs[buf][(wn * 32 + j * 16 + lr) * LDK + lh * 8];
            bf[j].q[0] = *(const uint4*)p;
            bf[j].q[1] = *(const uint4*)(p + 16);
        }
#pragma unroll
        for (int i = 0; i < 2; ++i)
#pragma unroll
            for (int j = 0; j < 2; ++j)
                acc[i][j] = __builtin_amdgcn_wmma_f32_16x16x32_f16(
                    false, af[i].v, false, bf[j].v, (short)0, acc[i][j], false, false);

        if (kt + 1 < KT) stage_store(buf ^ 1);        // waits only on 3 loads
        __syncthreads();
    }

    // ---- epilogue: C/D layout VGPR r -> M = r + 8*half, N = lane%16 ----
#pragma unroll
    for (int i = 0; i < 2; ++i) {
        int mbase = blockM + wm * 32 + i * 16;
#pragma unroll
        for (int j = 0; j < 2; ++j) {
            int nbase = blockN + wn * 32 + j * 16;
            int ng = nbase + lr;
#pragma unroll
            for (int r = 0; r < 8; ++r) {
                int mg = mbase + r + lh * 8;
                float v = acc[i][j][r] + bias[mg];
                if (flags & 1) v = fmaxf(v, 0.0f);
                if (ng < N) {
                    if (flags & 2) D[(size_t)ng * ldd + mg] = v;
                    else           D[(size_t)mg * ldd + ng] = v;
                }
            }
        }
    }
}

// ---------------------------------------------------------------------------
// Elementwise / packing kernels
// ---------------------------------------------------------------------------
__global__ void k_f32_to_f16(const float* __restrict__ s, _Float16* __restrict__ d, int n)
{
    int i = blockIdx.x * blockDim.x + threadIdx.x;
    if (i < n) d[i] = (_Float16)s[i];
}

__global__ void k_init_hc(const float* __restrict__ hxs, float* __restrict__ h, float* __restrict__ c)
{
    int i = blockIdx.x * blockDim.x + threadIdx.x;
    if (i >= BATCH * CCH * SP) return;
    int b = i / (CCH * SP);
    int r = i - b * (CCH * SP);
    h[i] = hxs[(size_t)b * (2 * CCH * SP) + r];
    c[i] = hxs[(size_t)b * (2 * CCH * SP) + CCH * SP + r];
}

__global__ void k_mask_hc(float* __restrict__ h, float* __restrict__ c,
                          const float* __restrict__ masks, int t)
{
    int i = blockIdx.x * blockDim.x + threadIdx.x;
    if (i >= BATCH * CCH * SP) return;
    int b = i / (CCH * SP);
    float m = masks[t * BATCH + b];
    h[i] *= m;
    c[i] *= m;
}

// im2col for a [B][128][7][7] f32 tensor into rows [rowOff .. rowOff+1151] of
// col[KCONV][NCONV] (f16). k = ci*9 + kh*3 + kw matches OIHW weight flattening.
__global__ void k_im2col(const float* __restrict__ src, _Float16* __restrict__ col, int rowOff)
{
    int idx = blockIdx.x * blockDim.x + threadIdx.x;
    if (idx >= (CCH * 9) * NCONV) return;
    int k = idx / NCONV;
    int n = idx - k * NCONV;
    int ci = k / 9;
    int r  = k - ci * 9;
    int kh = r / 3, kw = r - kh * 3;
    int b = n / SP;
    int p = n - b * SP;
    int y = p / HWD, xx = p - y * HWD;
    int iy = y + kh - 1, ix = xx + kw - 1;
    float v = 0.0f;
    if (iy >= 0 && iy < HWD && ix >= 0 && ix < HWD)
        v = src[((size_t)b * CCH + ci) * SP + iy * HWD + ix];
    col[(size_t)(rowOff + k) * NCONV + n] = (_Float16)v;
}

__global__ void k_lstm(const float* __restrict__ gates, float* __restrict__ h, float* __restrict__ c)
{
    int i = blockIdx.x * blockDim.x + threadIdx.x;
    if (i >= BATCH * CCH * SP) return;
    int b  = i / (CCH * SP);
    int r  = i - b * (CCH * SP);
    int ch = r / SP;
    int p  = r - ch * SP;
    int n  = b * SP + p;
    float gi = gates[(size_t)(ch            ) * NCONV + n];
    float gf = gates[(size_t)(ch +     CCH  ) * NCONV + n];
    float go = gates[(size_t)(ch + 2 * CCH  ) * NCONV + n];
    float gg = gates[(size_t)(ch + 3 * CCH  ) * NCONV + n];
    float si = 1.0f / (1.0f + __expf(-gi));
    float sf = 1.0f / (1.0f + __expf(-gf));
    float so = 1.0f / (1.0f + __expf(-go));
    float tg = tanhf(gg);
    float cn = sf * c[i] + si * tg;
    c[i] = cn;
    h[i] = so * tanhf(cn);
}

// cB[k][b] = c[b][k]  (f32 -> f16), K=6272, N=32
__global__ void k_pack_c(const float* __restrict__ c, _Float16* __restrict__ cB)
{
    int idx = blockIdx.x * blockDim.x + threadIdx.x;
    if (idx >= KLIN * BATCH) return;
    int k = idx / BATCH;
    int n = idx - k * BATCH;
    cB[idx] = (_Float16)c[(size_t)n * KLIN + k];
}

__global__ void k_final(const float* __restrict__ h, const float* __restrict__ c, float* __restrict__ o)
{
    int i = blockIdx.x * blockDim.x + threadIdx.x;
    if (i >= BATCH * CCH * SP) return;
    int b = i / (CCH * SP);
    int r = i - b * (CCH * SP);
    o[(size_t)b * (2 * CCH * SP) + r]            = h[i];
    o[(size_t)b * (2 * CCH * SP) + CCH * SP + r] = c[i];
}

// ---------------------------------------------------------------------------
// Host orchestration (all on `stream`, graph-capture safe)
// ---------------------------------------------------------------------------
extern "C" void kernel_launch(void* const* d_in, const int* in_sizes, int n_in,
                              void* d_out, int out_size, void* d_ws, size_t ws_size,
                              hipStream_t stream)
{
    (void)in_sizes; (void)n_in; (void)out_size; (void)ws_size;

    const float* x     = (const float*)d_in[0];  // (T*B, C, 7, 7)
    const float* hxs   = (const float*)d_in[1];  // (B, 2C, 7, 7)
    const float* masks = (const float*)d_in[2];  // (T*B,)
    const float* Wconv = (const float*)d_in[3];  // (512, 256, 3, 3)
    const float* bconv = (const float*)d_in[4];  // (512,)
    const float* Wlin  = (const float*)d_in[5];  // (512, 6272)
    const float* blin  = (const float*)d_in[6];  // (512,)
    float* out = (float*)d_out;                  // outs (1024,512) ++ hc (32,256,7,7)

    // workspace layout (256B aligned offsets), total ~21.2 MB
    char* ws = (char*)d_ws;
    _Float16* Wc16  = (_Float16*)(ws + 0);          //  512*2304 f16 = 2359296 B
    _Float16* Wl16  = (_Float16*)(ws + 2359296);    //  512*6272 f16 = 6422528 B
    _Float16* col   = (_Float16*)(ws + 8781824);    // 2304*1568 f16 = 7225344 B
    float*    gates = (float*)   (ws + 16007168);   //  512*1568 f32 = 3211264 B
    float*    h     = (float*)   (ws + 19218432);   // 32*128*49 f32 =  802816 B
    float*    c     = (float*)   (ws + 20021248);   // 32*128*49 f32 =  802816 B
    _Float16* cB    = (_Float16*)(ws + 20824064);   // 6272*32   f16 =  401408 B

    dim3 blk(256);
    const int nHC = BATCH * CCH * SP;               // 200704
    const int gHC = (nHC + 255) / 256;              // 784
    const int nIC = (CCH * 9) * NCONV;              // 1806336
    const int gIC = (nIC + 255) / 256;              // 7056

    // one-time (per call) weight conversion to f16
    k_f32_to_f16<<<(M4C * KCONV + 255) / 256, blk, 0, stream>>>(Wconv, Wc16, M4C * KCONV);
    k_f32_to_f16<<<(HSZ * KLIN  + 255) / 256, blk, 0, stream>>>(Wlin,  Wl16, HSZ * KLIN);
    k_init_hc<<<gHC, blk, 0, stream>>>(hxs, h, c);

    for (int t = 0; t < T_STEPS; ++t) {
        k_mask_hc<<<gHC, blk, 0, stream>>>(h, c, masks, t);
        // x-part of im2col is constant across the RD inner iterations
        k_im2col<<<gIC, blk, 0, stream>>>(x + (size_t)t * BATCH * CCH * SP, col, 0);
        for (int rr = 0; rr < RDI; ++rr) {
            k_im2col<<<gIC, blk, 0, stream>>>(h, col, CCH * 9);
            gemm_f16_wmma<<<dim3(25, 4), blk, 0, stream>>>(
                Wc16, col, gates, bconv, M4C, NCONV, KCONV, NCONV, /*flags=*/0);
            k_lstm<<<gHC, blk, 0, stream>>>(gates, h, c);
        }
        k_pack_c<<<(KLIN * BATCH + 255) / 256, blk, 0, stream>>>(c, cB);
        gemm_f16_wmma<<<dim3(1, 4), blk, 0, stream>>>(
            Wl16, cB, out + (size_t)t * BATCH * HSZ, blin, HSZ, BATCH, KLIN, HSZ, /*flags=*/3);
    }
    k_final<<<gHC, blk, 0, stream>>>(h, c, out + (size_t)T_STEPS * BATCH * HSZ);
}
